// np_ETE_58583353917524
// MI455X (gfx1250) — matrix operand order repaired and checked by reading, not compiled, not run on hardware
//
#include <hip/hip_runtime.h>

typedef __attribute__((ext_vector_type(16))) _Float16 v16h;
typedef __attribute__((ext_vector_type(8)))  _Float16 h8;
typedef __attribute__((ext_vector_type(4)))  _Float16 h4;
typedef __attribute__((ext_vector_type(8)))  float    v8f;
typedef int v4i __attribute__((vector_size(16)));

#define NDIM  8192
#define UNITS 128
#define BM    128
#define BN    128     // per-block columns of Wt (256 total via gridDim.x = 2)
#define LDSP  40      // padded LDS row stride in halves (80 B, conflict-free b128)
#define LBUF  (BM * LDSP)

#if defined(__has_builtin)
#if __has_builtin(__builtin_amdgcn_global_load_async_to_lds_b128) && \
    __has_builtin(__builtin_amdgcn_s_wait_asynccnt)
#define USE_ASYNC_LDS 1
#endif
#endif

#ifdef USE_ASYNC_LDS
__device__ __forceinline__ void async_copy16(const _Float16* g, _Float16* l) {
    __builtin_amdgcn_global_load_async_to_lds_b128(
        (__attribute__((address_space(1))) v4i*)(g),
        (__attribute__((address_space(3))) v4i*)(l), 0, 0);
}
#endif

// ---------------------------------------------------------------------------
// Kernel 1: rowsum / colsum of binary A in one streaming pass.
// Sums are integer-valued -> f32 atomics are exact and order-independent.
// ---------------------------------------------------------------------------
__global__ __launch_bounds__(256) void sums_kernel(const float* __restrict__ A,
                                                   float* __restrict__ rowsum,
                                                   float* __restrict__ colsum) {
    __shared__ float srow[64];
    const int t  = threadIdx.x;
    const int r0 = blockIdx.x * 64;
    if (t < 64) srow[t] = 0.f;
    __syncthreads();

    float ca[32];
#pragma unroll
    for (int i = 0; i < 32; ++i) ca[i] = 0.f;

    for (int r = 0; r < 64; ++r) {
        const float4* rowp = (const float4*)(A + (size_t)(r0 + r) * NDIM);
        float rp = 0.f;
#pragma unroll
        for (int i = 0; i < 8; ++i) {
            float4 v = rowp[t + 256 * i];
            ca[i * 4 + 0] += v.x; ca[i * 4 + 1] += v.y;
            ca[i * 4 + 2] += v.z; ca[i * 4 + 3] += v.w;
            rp += (v.x + v.y) + (v.z + v.w);
        }
#pragma unroll
        for (int o = 16; o > 0; o >>= 1) rp += __shfl_xor(rp, o, 32);
        if ((t & 31) == 0) atomicAdd(&srow[r], rp);
    }
    __syncthreads();
    if (t < 64) rowsum[r0 + t] = srow[t];

#pragma unroll
    for (int i = 0; i < 8; ++i)
#pragma unroll
        for (int c = 0; c < 4; ++c)
            atomicAdd(&colsum[1024 * i + 4 * t + c], ca[i * 4 + c]);
}

// ---------------------------------------------------------------------------
// Kernel 2: Wt[u][k] (f16, [256][8192]) = u<128 ? w[k][u] : colsum[k]*w[k][u-128]
// ---------------------------------------------------------------------------
__global__ __launch_bounds__(256) void wt_kernel(const float* __restrict__ w,
                                                 const float* __restrict__ colsum,
                                                 _Float16* __restrict__ Wt) {
    const int k  = blockIdx.x * 256 + threadIdx.x;
    const float cs = colsum[k];
#pragma unroll 4
    for (int u = 0; u < 256; ++u) {
        float wv = w[(size_t)k * UNITS + (u & 127)];
        if (u >= 128) wv *= cs;
        Wt[(size_t)u * NDIM + k] = (_Float16)wv;
    }
}

// ---------------------------------------------------------------------------
// Kernel 3: double-buffered, software-pipelined WMMA GEMM  C = f16(A) @ Wt^T
//   bx==0 (P part): out += (rowsum[i]-2)*C + b
//   bx==1 (Q part): out += C
// ---------------------------------------------------------------------------
__global__ __launch_bounds__(256) void gemm_kernel(const float* __restrict__ A,
                                                   const _Float16* __restrict__ Wt,
                                                   const float* __restrict__ rowsum,
                                                   const float* __restrict__ bias,
                                                   float* __restrict__ out) {
    __shared__ __align__(16) _Float16 As[2][LBUF];
    __shared__ __align__(16) _Float16 Bs[2][LBUF];

    const int t    = threadIdx.x;
    const int lane = t & 31;
    const int wid  = t >> 5;     // 0..7
    const int wm   = wid & 3;    // 32-row group within 128
    const int wn   = wid >> 2;   // 64-col group within 128
    const int lm   = lane & 15;
    const int half = lane >> 4;

    const int bx  = blockIdx.x;          // 0 = P, 1 = Q
    const int m0  = blockIdx.y * BM;
    const int nb0 = bx * BN;             // row offset into Wt's 256 rows

    // staging thread mappings
    const int ar = t >> 3;               // A: row 0..31 (+p*32)
    const int ac = (t & 7) * 4;          // A: col group (float4)
    const int bu = t >> 1;               // B: Wt row 0..127
    const int bc = (t & 1) * 16;         // B: col group (16 halves)

    const float*    Ab = A  + (size_t)(m0 + ar) * NDIM + ac;
    const _Float16* Wb = Wt + (size_t)(nb0 + bu) * NDIM + bc;

    v8f acc[2][4];
#pragma unroll
    for (int i = 0; i < 2; ++i)
#pragma unroll
        for (int j = 0; j < 4; ++j) acc[i][j] = {};

    float4 va[4];
#ifndef USE_ASYNC_LDS
    uint4 qb0, qb1;
#endif

    // ---- prologue: stage tile k0 = 0 into buffer 0
#pragma unroll
    for (int p = 0; p < 4; ++p)
        va[p] = *(const float4*)(Ab + (size_t)(p * 32) * NDIM);
#ifdef USE_ASYNC_LDS
    async_copy16(Wb,     &Bs[0][bu * LDSP + bc]);
    async_copy16(Wb + 8, &Bs[0][bu * LDSP + bc + 8]);
#else
    qb0 = *(const uint4*)(Wb);
    qb1 = *(const uint4*)(Wb + 8);
    *(uint4*)(&Bs[0][bu * LDSP + bc])     = qb0;
    *(uint4*)(&Bs[0][bu * LDSP + bc + 8]) = qb1;
#endif
#pragma unroll
    for (int p = 0; p < 4; ++p) {
        h4 hv;
        hv[0] = (_Float16)va[p].x; hv[1] = (_Float16)va[p].y;
        hv[2] = (_Float16)va[p].z; hv[3] = (_Float16)va[p].w;
        *(h4*)(&As[0][(p * 32 + ar) * LDSP + ac]) = hv;
    }
#ifdef USE_ASYNC_LDS
    __builtin_amdgcn_s_wait_asynccnt(0);
#endif
    __syncthreads();

    int buf = 0;
    for (int k0 = 0; k0 < NDIM; k0 += 32) {
        const bool more = (k0 + 32) < NDIM;
        const int  nxt  = buf ^ 1;

        // ---- issue next-tile global traffic first (overlaps the WMMAs)
        if (more) {
#pragma unroll
            for (int p = 0; p < 4; ++p) {
                const float* gp = Ab + (size_t)(p * 32) * NDIM + (k0 + 32);
                va[p] = *(const float4*)gp;
                __builtin_prefetch(gp + 32, 0, 0);   // speculative, OOB-safe
            }
#ifdef USE_ASYNC_LDS
            async_copy16(Wb + (k0 + 32),     &Bs[nxt][bu * LDSP + bc]);
            async_copy16(Wb + (k0 + 32) + 8, &Bs[nxt][bu * LDSP + bc + 8]);
#else
            qb0 = *(const uint4*)(Wb + (k0 + 32));
            qb1 = *(const uint4*)(Wb + (k0 + 32) + 8);
#endif
        }

        // ---- fragment loads per ISA 16-bit A(16x32) / B(32x16) lane layouts
        v16h af[2];
#pragma unroll
        for (int fm = 0; fm < 2; ++fm) {
            const int row = wm * 32 + fm * 16 + lm;
            h8 lo = *(const h8*)(&As[buf][row * LDSP + half * 8]);
            h8 hi = *(const h8*)(&As[buf][row * LDSP + 16 + half * 8]);
            af[fm] = __builtin_shufflevector(lo, hi,
                      0, 1, 2, 3, 4, 5, 6, 7, 8, 9, 10, 11, 12, 13, 14, 15);
        }
        v16h bf[4];
#pragma unroll
        for (int fn = 0; fn < 4; ++fn) {
            const int urow = wn * 64 + fn * 16 + lm;
            h8 lo = *(const h8*)(&Bs[buf][urow * LDSP + half * 8]);
            h8 hi = *(const h8*)(&Bs[buf][urow * LDSP + 16 + half * 8]);
            bf[fn] = __builtin_shufflevector(lo, hi,
                      0, 1, 2, 3, 4, 5, 6, 7, 8, 9, 10, 11, 12, 13, 14, 15);
        }

#pragma unroll
        for (int fm = 0; fm < 2; ++fm)
#pragma unroll
            for (int fn = 0; fn < 4; ++fn)
                acc[fm][fn] = __builtin_amdgcn_wmma_f32_16x16x32_f16(
                    false, af[fm], false, bf[fn],
                    (short)0, acc[fm][fn], false, false);

        // ---- drain next-tile staging into the other buffer
        if (more) {
#pragma unroll
            for (int p = 0; p < 4; ++p) {
                h4 hv;
                hv[0] = (_Float16)va[p].x; hv[1] = (_Float16)va[p].y;
                hv[2] = (_Float16)va[p].z; hv[3] = (_Float16)va[p].w;
                *(h4*)(&As[nxt][(p * 32 + ar) * LDSP + ac]) = hv;
            }
#ifndef USE_ASYNC_LDS
            *(uint4*)(&Bs[nxt][bu * LDSP + bc])     = qb0;
            *(uint4*)(&Bs[nxt][bu * LDSP + bc + 8]) = qb1;
#else
            __builtin_amdgcn_s_wait_asynccnt(0);
#endif
        }
        __syncthreads();
        buf = nxt;
    }

    // ---- fused epilogue (C layout: VGPR v -> M = v + 8*half, lane%16 -> N)
#pragma unroll
    for (int fm = 0; fm < 2; ++fm) {
#pragma unroll
        for (int fn = 0; fn < 4; ++fn) {
            const int uc = wn * 64 + fn * 16 + lm;   // output unit column 0..127
#pragma unroll
            for (int v = 0; v < 8; ++v) {
                const int row = m0 + wm * 32 + fm * 16 + half * 8 + v;
                float val = acc[fm][fn][v];
                if (bx == 0)
                    val = val * (rowsum[row] - 2.0f) + bias[uc];
                atomicAdd(&out[(size_t)row * UNITS + uc], val);
            }
        }
    }
}

// ---------------------------------------------------------------------------
extern "C" void kernel_launch(void* const* d_in, const int* in_sizes, int n_in,
                              void* d_out, int out_size, void* d_ws, size_t ws_size,
                              hipStream_t stream) {
    const float* A = (const float*)d_in[0];
    const float* w = (const float*)d_in[1];
    const float* b = (const float*)d_in[2];
    float* out = (float*)d_out;

    float*     rowsum = (float*)d_ws;                       // 8192 f32
    float*     colsum = rowsum + NDIM;                      // 8192 f32
    _Float16*  Wt     = (_Float16*)((char*)d_ws + 65536);   // 256*8192 f16 = 4 MB

    (void)hipMemsetAsync(colsum, 0, NDIM * sizeof(float), stream);
    (void)hipMemsetAsync(d_out, 0, (size_t)NDIM * UNITS * sizeof(float), stream);

    sums_kernel<<<NDIM / 64, 256, 0, stream>>>(A, rowsum, colsum);
    wt_kernel  <<<NDIM / 256, 256, 0, stream>>>(w, colsum, Wt);
    gemm_kernel<<<dim3(2, NDIM / BM), 256, 0, stream>>>(A, Wt, rowsum, b, out);
}
